// AutoEncoderTopK_34102040330297
// MI455X (gfx1250) — matrix operand order repaired and checked by reading, not compiled, not run on hardware
//
#include <hip/hip_runtime.h>
#include <hip/hip_bf16.h>

// Problem constants (match reference)
#define BB 4096
#define DD 1024
#define FF 32768
#define KK 64
#define CAP 4096   // LDS candidate buffer for top-k fast path

typedef __attribute__((ext_vector_type(16))) __bf16 v16bf;
typedef __attribute__((ext_vector_type(8)))  __bf16 v8bf;
typedef __attribute__((ext_vector_type(8)))  float  v8f;

// ---------------------------------------------------------------------------
// Kernel 1: xb = bf16(x - b_dec)
// ---------------------------------------------------------------------------
__global__ void k_conv_x(const float* __restrict__ x,
                         const float* __restrict__ b_dec,
                         __bf16* __restrict__ xb) {
    size_t i = (size_t)blockIdx.x * blockDim.x + threadIdx.x;
    int d = (int)(i & (DD - 1));
    xb[i] = (__bf16)(x[i] - b_dec[d]);
}

// ---------------------------------------------------------------------------
// Kernel 2: web = bf16(W_enc)   (F*D elements, 64 MB -> L2 resident)
// ---------------------------------------------------------------------------
__global__ void k_conv_we(const float* __restrict__ we,
                          __bf16* __restrict__ web) {
    size_t i = (size_t)blockIdx.x * blockDim.x + threadIdx.x;
    web[i] = (__bf16)we[i];
}

// ---------------------------------------------------------------------------
// Kernel 3: W_decT[f, d] = W_dec[d, f]   (tiled LDS transpose)
// ---------------------------------------------------------------------------
__global__ void k_transpose_wdec(const float* __restrict__ wdec,
                                 float* __restrict__ wdecT) {
    __shared__ float t[32][33];
    int f0 = blockIdx.x * 32;
    int d0 = blockIdx.y * 32;
    int tx = threadIdx.x;           // 0..31
    for (int j = threadIdx.y; j < 32; j += 8) {
        t[j][tx] = wdec[(size_t)(d0 + j) * FF + (f0 + tx)];
    }
    __syncthreads();
    for (int j = threadIdx.y; j < 32; j += 8) {
        wdecT[(size_t)(f0 + j) * DD + (d0 + tx)] = t[tx][j];
    }
}

// ---------------------------------------------------------------------------
// Kernel 4: encoder GEMM + bias + ReLU via v_wmma_f32_16x16x32_bf16
//   pre[b, f] = relu( sum_d xb[b,d] * web[f,d] + b_enc[f] )
//   Wave tile: 64 (rows) x 32 (cols) = 4x2 accumulators (A frags reused 2x,
//   B frags reused 4x). Block: 8 waves as 2(M) x 4(N) -> 128 x 128 tile.
//   W_enc (bf16, L2-resident) re-streamed only B/128 = 32 times (~2 GB L2).
// ---------------------------------------------------------------------------
__global__ __launch_bounds__(256)
void k_encoder_gemm(const __bf16* __restrict__ xb,
                    const __bf16* __restrict__ web,
                    const float* __restrict__ b_enc,
                    float* __restrict__ pre) {
    const int lane   = threadIdx.x & 31;
    const int wave   = threadIdx.x >> 5;
    const int wave_m = wave >> 2;   // 0..1
    const int wave_n = wave & 3;    // 0..3

    const int m_base = blockIdx.y * 128 + wave_m * 64;
    const int f_base = blockIdx.x * 128 + wave_n * 32;

    const int mrow = lane & 15;     // row-within-16-tile (A) / col-row (B)
    const int kh   = lane >> 4;     // which K-half this lane holds

    v8f acc[4][2] = {};

    const __bf16* arow0 = xb  + (size_t)(m_base + mrow) * DD;
    const __bf16* brow0 = web + (size_t)(f_base + mrow) * DD;

    for (int d0 = 0; d0 < DD; d0 += 32) {
        // A fragments (16x32 bf16 each): lanes 0-15 hold K=0..15,
        // lanes 16-31 hold K=16..31 of this 32-wide slab.
        v16bf a[4];
#pragma unroll
        for (int i = 0; i < 4; ++i) {
            const __bf16* ar = arow0 + (size_t)i * 16 * DD + d0 + kh * 16;
            v8bf lo = *(const v8bf*)ar;
            v8bf hi = *(const v8bf*)(ar + 8);
            a[i] = __builtin_shufflevector(lo, hi,
                     0, 1, 2, 3, 4, 5, 6, 7, 8, 9, 10, 11, 12, 13, 14, 15);
        }
        // B fragments (32x16 bf16 each): lane n=lane&15 is a W_enc row,
        // lanes 0-15 hold K=0..15, lanes 16-31 hold K=16..31 (32B load).
        v16bf b[2];
#pragma unroll
        for (int j = 0; j < 2; ++j) {
            b[j] = *(const v16bf*)(brow0 + (size_t)j * 16 * DD + d0 + kh * 16);
        }
#pragma unroll
        for (int i = 0; i < 4; ++i) {
#pragma unroll
            for (int j = 0; j < 2; ++j) {
                acc[i][j] = __builtin_amdgcn_wmma_f32_16x16x32_bf16(
                    false, a[i], false, b[j], (short)0, acc[i][j], false, false);
            }
        }
    }

    // C/D layout: VGPR r -> M = r + 8*(lane>=16), N = lane&15
    const int n_out = lane & 15;
    const int mhi   = (lane >> 4) * 8;

#pragma unroll
    for (int j = 0; j < 2; ++j) {
        const int f = f_base + j * 16 + n_out;
        const float be = b_enc[f];
#pragma unroll
        for (int i = 0; i < 4; ++i) {
#pragma unroll
            for (int r = 0; r < 8; ++r) {
                float v = acc[i][j][r] + be;
                v = v > 0.0f ? v : 0.0f;
                pre[(size_t)(m_base + i * 16 + mhi + r) * FF + f] = v;
            }
        }
    }
}

// ---------------------------------------------------------------------------
// Kernel 5: per-row top-K.
//   pre >= 0 (post-ReLU) => raw float bits are a monotonic unsigned key.
//   Fast path (2 global scans): byte histogram -> compact candidates
//   (topbyte >= threshold byte) into LDS -> finish radix select in LDS.
//   Fallback (rare, e.g. huge tie bins): full global radix on lower bytes.
// ---------------------------------------------------------------------------
__global__ __launch_bounds__(256)
void k_topk(const float* __restrict__ pre,
            float* __restrict__ vals,
            int* __restrict__ idx) {
    const int tid = threadIdx.x;
    const float* row = pre + (size_t)blockIdx.x * FF;
    const size_t base = (size_t)blockIdx.x * KK;

    __shared__ unsigned sh_hist[256];
    __shared__ float sh_cv[CAP];
    __shared__ int   sh_ci[CAP];
    __shared__ int sh_d8, sh_numAbove, sh_cand;
    __shared__ unsigned sh_prefix, sh_mask;
    __shared__ int sh_remaining;
    __shared__ int sh_cnt, sh_cHi, sh_cTie;

    // ---- scan 1: top-byte histogram ----
    sh_hist[tid] = 0u;
    __syncthreads();
    for (int i = tid; i < FF; i += 256) {
        unsigned u = __float_as_uint(row[i]);
        atomicAdd(&sh_hist[u >> 24], 1u);
    }
    __syncthreads();
    if (tid == 0) {
        unsigned cum = 0;
        int d = 255;
        for (; d >= 0; --d) {
            unsigned c = sh_hist[d];
            if (cum + c >= (unsigned)KK) break;
            cum += c;
        }
        if (d < 0) d = 0;
        sh_d8 = d;
        sh_numAbove = (int)cum;                 // elems strictly above bin d
        sh_cand = (int)(cum + sh_hist[d]);      // candidates (byte >= d)
        sh_cnt = 0; sh_cHi = 0; sh_cTie = 0;
    }
    __syncthreads();

    const int d8 = sh_d8;
    const int numAbove = sh_numAbove;
    const int cand = sh_cand;

    if (cand <= CAP) {
        // ---- scan 2: compact candidates into LDS ----
        for (int i = tid; i < FF; i += 256) {
            float v = row[i];
            unsigned u = __float_as_uint(v);
            if ((int)(u >> 24) >= d8) {
                int p = atomicAdd(&sh_cnt, 1);
                sh_cv[p] = v;
                sh_ci[p] = i;
            }
        }
        if (tid == 0) {
            sh_prefix = (unsigned)d8 << 24;
            sh_mask = 0xFF000000u;
            sh_remaining = KK - numAbove;
        }
        __syncthreads();

        // ---- radix select lower 3 bytes, entirely in LDS ----
        for (int pass = 1; pass < 4; ++pass) {
            const int shift = 24 - 8 * pass;
            sh_hist[tid] = 0u;
            __syncthreads();
            const unsigned pfx = sh_prefix, msk = sh_mask;
            for (int p = tid; p < cand; p += 256) {
                unsigned u = __float_as_uint(sh_cv[p]);
                if ((u & msk) == pfx)
                    atomicAdd(&sh_hist[(u >> shift) & 255], 1u);
            }
            __syncthreads();
            if (tid == 0) {
                int rem = sh_remaining;
                unsigned cum = 0;
                int d = 255;
                for (; d >= 0; --d) {
                    unsigned c = sh_hist[d];
                    if (cum + c >= (unsigned)rem) break;
                    cum += c;
                }
                if (d < 0) d = 0;
                sh_remaining = rem - (int)cum;
                sh_prefix = pfx | ((unsigned)d << shift);
                sh_mask = msk | (0xFFu << shift);
            }
            __syncthreads();
        }
        const unsigned T = sh_prefix;
        const int needTies = sh_remaining;
        const int numHi = KK - needTies;

        for (int p = tid; p < cand; p += 256) {
            float v = sh_cv[p];
            unsigned u = __float_as_uint(v);
            if (u > T) {
                int s = atomicAdd(&sh_cHi, 1);
                vals[base + s] = v;
                idx[base + s] = sh_ci[p];
            } else if (u == T) {
                int s = atomicAdd(&sh_cTie, 1);
                if (s < needTies) {
                    vals[base + numHi + s] = v;
                    idx[base + numHi + s] = sh_ci[p];
                }
            }
        }
    } else {
        // ---- fallback: global radix on lower bytes + global collection ----
        if (tid == 0) {
            sh_prefix = (unsigned)d8 << 24;
            sh_mask = 0xFF000000u;
            sh_remaining = KK - numAbove;
        }
        __syncthreads();
        for (int pass = 1; pass < 4; ++pass) {
            const int shift = 24 - 8 * pass;
            sh_hist[tid] = 0u;
            __syncthreads();
            const unsigned pfx = sh_prefix, msk = sh_mask;
            for (int i = tid; i < FF; i += 256) {
                unsigned u = __float_as_uint(row[i]);
                if ((u & msk) == pfx)
                    atomicAdd(&sh_hist[(u >> shift) & 255], 1u);
            }
            __syncthreads();
            if (tid == 0) {
                int rem = sh_remaining;
                unsigned cum = 0;
                int d = 255;
                for (; d >= 0; --d) {
                    unsigned c = sh_hist[d];
                    if (cum + c >= (unsigned)rem) break;
                    cum += c;
                }
                if (d < 0) d = 0;
                sh_remaining = rem - (int)cum;
                sh_prefix = pfx | ((unsigned)d << shift);
                sh_mask = msk | (0xFFu << shift);
            }
            __syncthreads();
        }
        const unsigned T = sh_prefix;
        const int needTies = sh_remaining;
        const int numHi = KK - needTies;

        for (int i = tid; i < FF; i += 256) {
            float v = row[i];
            unsigned u = __float_as_uint(v);
            if (u > T) {
                int s = atomicAdd(&sh_cHi, 1);
                vals[base + s] = v;
                idx[base + s] = i;
            } else if (u == T) {
                int s = atomicAdd(&sh_cTie, 1);
                if (s < needTies) {
                    vals[base + numHi + s] = v;
                    idx[base + numHi + s] = i;
                }
            }
        }
    }
}

// ---------------------------------------------------------------------------
// Kernel 6: sparse decode
//   out[b, d] = b_dec[d] + sum_k vals[b,k] * W_decT[idx[b,k], d]
// ---------------------------------------------------------------------------
__global__ __launch_bounds__(256)
void k_decode(const float* __restrict__ vals,
              const int* __restrict__ idx,
              const float* __restrict__ wdecT,
              const float* __restrict__ b_dec,
              float* __restrict__ out) {
    const int tid = threadIdx.x;
    const size_t b = blockIdx.x;

    __shared__ float sval[KK];
    __shared__ int   sidx[KK];
    if (tid < KK) {
        sval[tid] = vals[b * KK + tid];
        sidx[tid] = idx[b * KK + tid];
    }
    __syncthreads();

    for (int d = tid; d < DD; d += 256) {
        float acc = b_dec[d];
#pragma unroll 8
        for (int k = 0; k < KK; ++k) {
            acc += sval[k] * wdecT[(size_t)sidx[k] * DD + d];
        }
        out[b * DD + d] = acc;
    }
}

// ---------------------------------------------------------------------------
// Host launcher
// ---------------------------------------------------------------------------
extern "C" void kernel_launch(void* const* d_in, const int* in_sizes, int n_in,
                              void* d_out, int out_size, void* d_ws, size_t ws_size,
                              hipStream_t stream) {
    (void)in_sizes; (void)n_in; (void)out_size; (void)ws_size;

    const float* x     = (const float*)d_in[0];   // [B, D]
    const float* W_enc = (const float*)d_in[1];   // [F, D]
    const float* b_enc = (const float*)d_in[2];   // [F]
    const float* W_dec = (const float*)d_in[3];   // [D, F]
    const float* b_dec = (const float*)d_in[4];   // [D]
    float* out = (float*)d_out;                   // [B, D]

    // Workspace layout (bytes)
    char* ws = (char*)d_ws;
    size_t off = 0;
    float*  pre   = (float*)(ws + off);  off += (size_t)BB * FF * sizeof(float);   // 512 MB
    __bf16* xb    = (__bf16*)(ws + off); off += (size_t)BB * DD * sizeof(__bf16);  // 8 MB
    __bf16* web   = (__bf16*)(ws + off); off += (size_t)FF * DD * sizeof(__bf16);  // 64 MB
    float*  wdecT = (float*)(ws + off);  off += (size_t)FF * DD * sizeof(float);   // 128 MB
    float*  tvals = (float*)(ws + off);  off += (size_t)BB * KK * sizeof(float);   // 1 MB
    int*    tidx  = (int*)(ws + off);    off += (size_t)BB * KK * sizeof(int);     // 1 MB

    // 1) x - b_dec -> bf16
    k_conv_x<<<(BB * DD) / 256, 256, 0, stream>>>(x, b_dec, xb);
    // 2) W_enc -> bf16 (64 MB, lives in L2 across the GEMM)
    k_conv_we<<<((size_t)FF * DD) / 256, 256, 0, stream>>>(W_enc, web);
    // 3) W_dec transpose for contiguous decode gathers
    k_transpose_wdec<<<dim3(FF / 32, DD / 32), dim3(32, 8), 0, stream>>>(W_dec, wdecT);
    // 4) encoder GEMM (WMMA bf16 -> f32) + bias + ReLU, 128x128 block tiles
    k_encoder_gemm<<<dim3(FF / 128, BB / 128), 256, 0, stream>>>(xb, web, b_enc, pre);
    // 5) per-row top-64 (2-scan histogram select, LDS finish)
    k_topk<<<BB, 256, 0, stream>>>(pre, tvals, tidx);
    // 6) sparse decode
    k_decode<<<BB, 256, 0, stream>>>(tvals, tidx, wdecT, b_dec, out);
}
